// APPNP_Net_47107201303142
// MI455X (gfx1250) — matrix operand order repaired and checked by reading, not compiled, not run on hardware
//
#include <hip/hip_runtime.h>
#include <stdint.h>

// ---------------------------------------------------------------------------
// APPNP on MI455X (gfx1250, wave32):
//   GEMM1 (512->256) via v_wmma_f32_16x16x32_f16 with vectorized f32->f16
//   staging (x streamed once at HBM rate), W1 pre-converted to f16 [n][k],
//   GEMM2 (256->32) scalar FMA, 10 propagation steps (self-loop folded into
//   the init pass, f32 atomics for the 1.6M-edge scatter, all hot data in L2).
// ---------------------------------------------------------------------------

typedef __attribute__((ext_vector_type(16))) _Float16 v16h;
typedef __attribute__((ext_vector_type(8)))  _Float16 v8h;
typedef __attribute__((ext_vector_type(4)))  _Float16 v4h;
typedef __attribute__((ext_vector_type(8)))  float    v8f;

#define IN_CH  512
#define HID_CH 256
#define OUT_CH 32
#define KSTEPS 10
#define ALPHA  0.1f
#define BETA   0.9f   // 1 - ALPHA

// ---------------------------------------------------------------------------
// One-time: W1[512,256] f32 -> W1h[n][k] f16 (transposed, 256 KB).
// ---------------------------------------------------------------------------
__global__ __launch_bounds__(256) void w1_to_f16_T(
    const float* __restrict__ W1, _Float16* __restrict__ W1h)
{
    int i = blockIdx.x * 256 + threadIdx.x;       // over 512*256, coalesced read
    int k = i >> 8, n = i & 255;
    W1h[(size_t)n * IN_CH + k] = (_Float16)W1[i];
}

// ---------------------------------------------------------------------------
// GEMM1 + bias + ReLU:  h1[N,256] = relu(x[N,512] @ W1[512,256] + b1)
// Block = 128 threads (4 waves). Block tile = 64 rows x 64 cols.
// Wave w computes rows [w*16, w*16+16) x 64 cols -> 4 accumulators (v8f).
// ---------------------------------------------------------------------------
__global__ __launch_bounds__(128) void gemm1_relu_wmma(
    const float* __restrict__ x, const _Float16* __restrict__ W1h,
    const float* __restrict__ b1, float* __restrict__ h1, int N)
{
    __shared__ _Float16 ldsA[64 * 32];   // [row][k]  4 KB
    __shared__ _Float16 ldsB[64 * 32];   // [col][k]  4 KB

    const int tid  = threadIdx.x;
    const int wave = tid >> 5;
    const int lane = tid & 31;
    const int hi   = lane >> 4;          // K-half select (wave32 lane groups)
    const int l16  = lane & 15;
    const int row0 = blockIdx.x * 64;
    const int n0   = blockIdx.y * 64;
    const bool full = (row0 + 64) <= N;  // all but the last row-block

    v8f acc[4];
    #pragma unroll
    for (int ct = 0; ct < 4; ++ct) acc[ct] = (v8f){0.f,0.f,0.f,0.f,0.f,0.f,0.f,0.f};

    for (int k0 = 0; k0 < IN_CH; k0 += 32) {
        // ---- stage A tile: x[row0..+63][k0..+31], f32 -> f16 ----
        if (full) {
            // 2048 floats = 128 threads * 4 float4; batch loads, then cvt+store
            float4 tmp[4];
            #pragma unroll
            for (int s = 0; s < 4; ++s) {
                int i = tid + s * 128;              // 512 float4 slots
                int r = i >> 3, kq = i & 7;
                tmp[s] = *(const float4*)(x + (size_t)(row0 + r) * IN_CH + k0 + kq * 4);
            }
            if (k0 + 32 < IN_CH)                    // prefetch next A k-tile
                __builtin_prefetch(x + (size_t)(row0 + (tid >> 3)) * IN_CH
                                     + (k0 + 32) + (tid & 7) * 4, 0, 0);
            #pragma unroll
            for (int s = 0; s < 4; ++s) {
                int i = tid + s * 128;
                int r = i >> 3, kq = i & 7;
                v4h h4;
                h4[0] = (_Float16)tmp[s].x;
                h4[1] = (_Float16)tmp[s].y;
                h4[2] = (_Float16)tmp[s].z;
                h4[3] = (_Float16)tmp[s].w;
                *(v4h*)&ldsA[r * 32 + kq * 4] = h4;   // ds_store_b64
            }
        } else {
            #pragma unroll
            for (int s = 0; s < 16; ++s) {
                int i = tid + s * 128;
                int r = i >> 5, k = i & 31;
                int row = row0 + r;
                float v = (row < N) ? x[(size_t)row * IN_CH + k0 + k] : 0.0f;
                ldsA[r * 32 + k] = (_Float16)v;
            }
        }
        // ---- stage B tile from pre-converted W1h[n][k]: pure 16B copies ----
        #pragma unroll
        for (int s = 0; s < 2; ++s) {
            int i = tid + s * 128;                  // 256 v8h chunks
            int n = i >> 2, ko = (i & 3) * 8;
            *(v8h*)&ldsB[n * 32 + ko] =
                *(const v8h*)(W1h + (size_t)(n0 + n) * IN_CH + k0 + ko);
        }
        __syncthreads();

        // A fragment: 16x32 f16, this code's contiguous k order:
        //   lanes 0-15 = rows 0-15; element j: K = 16*hi + j (matches B)
        v16h a;
        {
            const _Float16* Ar = &ldsA[(wave * 16 + l16) * 32];
            #pragma unroll
            for (int j = 0; j < 16; ++j) {
                a[j]     = Ar[16 * hi + j];
            }
        }

        #pragma unroll
        for (int ct = 0; ct < 4; ++ct) {
            // B fragment: 32x16 — lanes 0-15 hold K=0..15, lanes 16-31 hold
            // K=16..31; element j <-> K = 16*hi + j (contiguous in ldsB[n][k]).
            v16h b;
            const _Float16* Bc = &ldsB[(ct * 16 + l16) * 32 + 16 * hi];
            #pragma unroll
            for (int j = 0; j < 16; ++j) b[j] = Bc[j];

            acc[ct] = __builtin_amdgcn_wmma_f32_16x16x32_f16(
                false, a, false, b, (short)0, acc[ct], false, false);
        }
        __syncthreads();
    }

    // Epilogue: C layout — lane n=l16, VGPR r -> row (r + 8*hi)
    #pragma unroll
    for (int ct = 0; ct < 4; ++ct) {
        int col  = n0 + ct * 16 + l16;
        float bv = b1[col];
        #pragma unroll
        for (int r = 0; r < 8; ++r) {
            int row = row0 + wave * 16 + hi * 8 + r;
            if (row < N) {
                float v = acc[ct][r] + bv;
                h1[(size_t)row * HID_CH + col] = v > 0.0f ? v : 0.0f;
            }
        }
    }
}

// ---------------------------------------------------------------------------
// GEMM2 + bias:  h0[N,32] = h1[N,256] @ W2[256,32] + b2
// Block 256 threads = 8 rows x 32 cols; W2 staged in LDS (32 KB).
// ---------------------------------------------------------------------------
__global__ __launch_bounds__(256) void gemm2_bias(
    const float* __restrict__ h1, const float* __restrict__ W2,
    const float* __restrict__ b2, float* __restrict__ h0, int N)
{
    __shared__ float w2s[HID_CH * OUT_CH];
    const int tid = threadIdx.x;
    for (int i = tid; i < HID_CH * OUT_CH; i += 256) w2s[i] = W2[i];
    __syncthreads();

    const int r   = tid >> 5;
    const int c   = tid & 31;
    const int row = blockIdx.x * 8 + r;
    if (row >= N) return;

    float acc = b2[c];
    const float* hr = &h1[(size_t)row * HID_CH];
    #pragma unroll 8
    for (int k = 0; k < HID_CH; ++k)
        acc = fmaf(hr[k], w2s[k * OUT_CH + c], acc);
    h0[(size_t)row * OUT_CH + c] = acc;
}

// ---------------------------------------------------------------------------
// Degree (with self-loop) and dinv = rsqrt(deg)
// ---------------------------------------------------------------------------
__global__ void deg_fill(float* __restrict__ deg, int N)
{
    int i = blockIdx.x * 256 + threadIdx.x;
    if (i < N) deg[i] = 1.0f;   // self-loop contribution
}

__global__ void deg_count(const long long* __restrict__ ei,
                          float* __restrict__ deg, int E)
{
    int e = blockIdx.x * 256 + threadIdx.x;
    if (e < E) atomicAdd(&deg[(int)ei[(size_t)E + e]], 1.0f);  // dst row
}

__global__ void deg_rsqrt(float* __restrict__ deg, int N)
{
    int i = blockIdx.x * 256 + threadIdx.x;
    if (i < N) deg[i] = rsqrtf(deg[i]);
}

// ---------------------------------------------------------------------------
// Propagation step, part 1: teleport + self-loop term (fully writes hnext)
//   hnext = ALPHA*h0 + BETA*dinv^2*hcur
// ---------------------------------------------------------------------------
__global__ __launch_bounds__(256) void prop_init(
    const float* __restrict__ h0, const float* __restrict__ hcur,
    const float* __restrict__ dinv, float* __restrict__ hnext, int N)
{
    int i = blockIdx.x * 256 + threadIdx.x;           // over N*32
    if (i < N * OUT_CH) {
        float di = dinv[i >> 5];
        hnext[i] = ALPHA * h0[i] + (BETA * di * di) * hcur[i];
    }
}

// ---------------------------------------------------------------------------
// Propagation step, part 2: edge scatter. One edge per 32-lane group,
// lane = channel -> 128B coalesced gather + 128B f32-atomic scatter.
// h (12.8 MB) + edges (25.6 MB) + dinv live in the 192 MB L2.
// ---------------------------------------------------------------------------
__global__ __launch_bounds__(256) void prop_edges(
    const long long* __restrict__ ei, const float* __restrict__ dinv,
    const float* __restrict__ hcur, float* __restrict__ hnext, int E)
{
    int e = blockIdx.x * 8 + (threadIdx.x >> 5);
    int c = threadIdx.x & 31;
    if (e >= E) return;
    int s = (int)ei[e];
    int d = (int)ei[(size_t)E + e];
    float w = BETA * dinv[s] * dinv[d];
    atomicAdd(&hnext[(size_t)d * OUT_CH + c], w * hcur[(size_t)s * OUT_CH + c]);
}

// ---------------------------------------------------------------------------
extern "C" void kernel_launch(void* const* d_in, const int* in_sizes, int n_in,
                              void* d_out, int out_size, void* d_ws, size_t ws_size,
                              hipStream_t stream)
{
    const float*     x  = (const float*)d_in[0];
    const long long* ei = (const long long*)d_in[1];   // int64 [2, E]
    const float*     W1 = (const float*)d_in[2];
    const float*     b1 = (const float*)d_in[3];
    const float*     W2 = (const float*)d_in[4];
    const float*     b2 = (const float*)d_in[5];
    float*           out = (float*)d_out;

    const int N = in_sizes[0] / IN_CH;
    const int E = in_sizes[1] / 2;

    // Workspace layout (all regions fully rewritten each call):
    //   [0, h1Bytes)              : h1 [N,256] -- dead after gemm2; hA/hB alias
    //   [h1Bytes, +hBytes)        : h0 [N,32]
    //   [.., +N*4)                : dinv [N]
    //   [.., +512*256*2)          : W1h f16 transposed
    char* ws = (char*)d_ws;
    const size_t h1Bytes = ((size_t)N * HID_CH * sizeof(float) + 255) & ~(size_t)255;
    const size_t hBytes  = ((size_t)N * OUT_CH * sizeof(float) + 255) & ~(size_t)255;
    const size_t dBytes  = ((size_t)N * sizeof(float) + 255) & ~(size_t)255;
    float*     h1   = (float*)ws;
    float*     hA   = (float*)ws;                 // aliases h1 (safe post-gemm2)
    float*     hB   = (float*)(ws + hBytes);      // also inside dead h1 region
    float*     h0   = (float*)(ws + h1Bytes);
    float*     dinv = (float*)(ws + h1Bytes + hBytes);
    _Float16*  W1h  = (_Float16*)(ws + h1Bytes + hBytes + dBytes);

    // MLP
    w1_to_f16_T<<<(IN_CH * HID_CH) / 256, 256, 0, stream>>>(W1, W1h);
    dim3 g1((N + 63) / 64, HID_CH / 64);
    gemm1_relu_wmma<<<g1, 128, 0, stream>>>(x, W1h, b1, h1, N);
    gemm2_bias<<<(N + 7) / 8, 256, 0, stream>>>(h1, W2, b2, h0, N);

    // GCN normalization
    deg_fill <<<(N + 255) / 256, 256, 0, stream>>>(dinv, N);
    deg_count<<<(E + 255) / 256, 256, 0, stream>>>(ei, dinv, E);
    deg_rsqrt<<<(N + 255) / 256, 256, 0, stream>>>(dinv, N);

    // K propagation steps; final step writes d_out directly.
    const float* cur = h0;
    for (int t = 0; t < KSTEPS; ++t) {
        float* nxt = (t == KSTEPS - 1) ? out : ((t & 1) ? hB : hA);
        prop_init <<<(N * OUT_CH + 255) / 256, 256, 0, stream>>>(h0, cur, dinv, nxt, N);
        prop_edges<<<(E + 7) / 8, 256, 0, stream>>>(ei, dinv, cur, nxt, E);
        cur = nxt;
    }
}
